// EncoderLayer_28724741276304
// MI455X (gfx1250) — matrix operand order repaired and checked by reading, not compiled, not run on hardware
//
#include <hip/hip_runtime.h>
#include <hip/hip_bf16.h>

typedef __attribute__((ext_vector_type(16))) __bf16 v16bf;
typedef __attribute__((ext_vector_type(8)))  __bf16 v8bf;
typedef __attribute__((ext_vector_type(8)))  float  v8f;

#define D_    1024
#define H_    16
#define DH_   64
#define F_    4096
#define B_    4
#define L_    2048
#define MROWS (B_ * L_)        // 8192
#define WPB   8                // waves per block (256 threads, wave32)

// ---------------------------------------------------------------- convert
__global__ __launch_bounds__(256) void cvt_f32_bf16(const float* __restrict__ in,
                                                    __bf16* __restrict__ out,
                                                    size_t n) {
    size_t i = (size_t)blockIdx.x * blockDim.x + threadIdx.x;
    size_t stride = (size_t)gridDim.x * blockDim.x;
    for (; i < n; i += stride) out[i] = (__bf16)in[i];
}

// ---------------------------------------------------------------- GEMM
// D[M,N] = A[M,K] * W[N,K]^T  (+bias[N]) (+relu) (+resid[M,N])
// One wave computes a 64x64 tile: 4x4 fragments of 16x16, K stepped by 32.
// Fragment layouts match CDNA5 ISA 7.12.2 (16-bit A 16x32, B 32x16).
__global__ __launch_bounds__(256) void gemm_bf16_wmma(
    const __bf16* __restrict__ A, const __bf16* __restrict__ W,
    const float* __restrict__ bias, const float* __restrict__ resid,
    float* __restrict__ outF, __bf16* __restrict__ outB,
    int M, int N, int K,
    long long batchA, long long batchW, long long batchOut,
    int relu)
{
    const int wave = threadIdx.x >> 5;
    const int lane = threadIdx.x & 31;
    const int half = lane >> 4;    // 0: lanes 0-15, 1: lanes 16-31
    const int r16  = lane & 15;

    const int tilesN = N >> 6;
    const int tile = blockIdx.x * WPB + wave;
    const int tm = tile / tilesN;
    const int tn = tile % tilesN;
    const int b  = blockIdx.y;

    const __bf16* Ab = A + (size_t)b * batchA;
    const __bf16* Wb = W + (size_t)b * batchW;

    const int am = tm * 64;
    const int wn = tn * 64;

    v8f acc[4][4];
#pragma unroll
    for (int mi = 0; mi < 4; ++mi)
#pragma unroll
        for (int ni = 0; ni < 4; ++ni)
#pragma unroll
            for (int r = 0; r < 8; ++r) acc[mi][ni][r] = 0.f;

    for (int k0 = 0; k0 < K; k0 += 32) {
        v16bf af[4], bf[4];
        // A fragment: lane holds row (am+mi*16+r16); elems 0..7 = K[k0+half*8 ..],
        // elems 8..15 = K[k0+16+half*8 ..]  (two contiguous b128 loads)
#pragma unroll
        for (int mi = 0; mi < 4; ++mi) {
            const __bf16* p = Ab + (size_t)(am + mi * 16 + r16) * K + k0 + half * 8;
            union { v16bf v; v8bf h[2]; } u;
            u.h[0] = *(const v8bf*)(p);
            u.h[1] = *(const v8bf*)(p + 16);
            af[mi] = u.v;
        }
        // B fragment: B[k][n] = W[n][k]; lane holds column (wn+ni*16+r16),
        // elems 0..7 = K[k0+half*8 ..], elems 8..15 = K[k0+16+half*8 ..] (two b128 loads)
#pragma unroll
        for (int ni = 0; ni < 4; ++ni) {
            const __bf16* p = Wb + (size_t)(wn + ni * 16 + r16) * K + k0 + half * 8;
            union { v16bf v; v8bf h[2]; } u;
            u.h[0] = *(const v8bf*)(p);
            u.h[1] = *(const v8bf*)(p + 16);
            bf[ni] = u.v;
        }
#pragma unroll
        for (int mi = 0; mi < 4; ++mi)
#pragma unroll
            for (int ni = 0; ni < 4; ++ni)
                acc[mi][ni] = __builtin_amdgcn_wmma_f32_16x16x32_bf16(
                    false, af[mi], false, bf[ni], (short)0, acc[mi][ni], false, false);
    }

    // Epilogue. C/D layout: lanes 0-15: vgpr r -> (M=r, N=lane); lanes 16-31: (M=8+r, N=lane-16)
#pragma unroll
    for (int mi = 0; mi < 4; ++mi) {
#pragma unroll
        for (int ni = 0; ni < 4; ++ni) {
            const int col = wn + ni * 16 + r16;
            const float bv = bias ? bias[col] : 0.f;
#pragma unroll
            for (int r = 0; r < 8; ++r) {
                const int row = am + mi * 16 + half * 8 + r;
                const size_t idx = (size_t)row * N + col;
                float v = acc[mi][ni][r] + bv;
                if (relu) v = v > 0.f ? v : 0.f;
                if (resid) v += resid[idx];
                if (outF) outF[(size_t)b * batchOut + idx] = v;
                if (outB) outB[idx] = (__bf16)v;
            }
        }
    }
}

// ---------------------------------------------------------------- softmax rows of 2048 (in place)
// logits scaled by DH^-0.5 = 0.125 before softmax; result divided by H.
__global__ __launch_bounds__(256) void softmax2048(float* __restrict__ s) {
    __shared__ float red[256];
    float* p = s + (size_t)blockIdx.x * 2048;
    const int t = threadIdx.x;
    float v[8];
    float mx = -3.4e38f;
#pragma unroll
    for (int i = 0; i < 8; ++i) {
        float x = p[t + i * 256] * 0.125f;
        v[i] = x;
        mx = fmaxf(mx, x);
    }
    red[t] = mx; __syncthreads();
    for (int o = 128; o > 0; o >>= 1) { if (t < o) red[t] = fmaxf(red[t], red[t + o]); __syncthreads(); }
    mx = red[0]; __syncthreads();
    float sum = 0.f;
#pragma unroll
    for (int i = 0; i < 8; ++i) { v[i] = __expf(v[i] - mx); sum += v[i]; }
    red[t] = sum; __syncthreads();
    for (int o = 128; o > 0; o >>= 1) { if (t < o) red[t] += red[t + o]; __syncthreads(); }
    const float inv = 1.f / (red[0] * (float)H_);
#pragma unroll
    for (int i = 0; i < 8; ++i) p[t + i * 256] = v[i] * inv;
}

// ---------------------------------------------------------------- LayerNorm rows of 1024
__global__ __launch_bounds__(256) void layernorm1024(const float* __restrict__ y,
                                                     const float* __restrict__ g,
                                                     const float* __restrict__ bt,
                                                     float* __restrict__ out) {
    __shared__ float rs_[256];
    __shared__ float rq_[256];
    const float* p = y + (size_t)blockIdx.x * 1024;
    float* o = out + (size_t)blockIdx.x * 1024;
    const int t = threadIdx.x;
    float v[4], s = 0.f, s2 = 0.f;
#pragma unroll
    for (int i = 0; i < 4; ++i) {
        float x = p[t + i * 256];
        v[i] = x; s += x; s2 += x * x;
    }
    rs_[t] = s; rq_[t] = s2; __syncthreads();
    for (int o2 = 128; o2 > 0; o2 >>= 1) {
        if (t < o2) { rs_[t] += rs_[t + o2]; rq_[t] += rq_[t + o2]; }
        __syncthreads();
    }
    const float mean = rs_[0] * (1.f / 1024.f);
    const float var  = rq_[0] * (1.f / 1024.f) - mean * mean;
    const float rstd = rsqrtf(var + 1e-5f);
#pragma unroll
    for (int i = 0; i < 4; ++i) {
        const int c = t + i * 256;
        o[c] = (v[i] - mean) * rstd * g[c] + bt[c];
    }
}

// ---------------------------------------------------------------- host
extern "C" void kernel_launch(void* const* d_in, const int* in_sizes, int n_in,
                              void* d_out, int out_size, void* d_ws, size_t ws_size,
                              hipStream_t stream) {
    (void)in_sizes; (void)n_in; (void)out_size; (void)ws_size;
    const float* x   = (const float*)d_in[0];
    const float* Wq  = (const float*)d_in[1];
    const float* Wk  = (const float*)d_in[2];
    const float* Wv  = (const float*)d_in[3];
    const float* Wo  = (const float*)d_in[4];
    const float* W1  = (const float*)d_in[5];
    const float* b1  = (const float*)d_in[6];
    const float* W2  = (const float*)d_in[7];
    const float* b2  = (const float*)d_in[8];
    const float* lng = (const float*)d_in[9];
    const float* lnb = (const float*)d_in[10];

    float* out  = (float*)d_out;                          // (B,L,D)
    float* attn = out + (size_t)MROWS * D_;               // (B,L,L)

    // workspace carve-out (256B aligned)
    char* base = (char*)d_ws;
    size_t off = 0;
    auto alloc = [&](size_t bytes) -> char* {
        off = (off + 255) & ~(size_t)255;
        char* p = base + off;
        off += bytes;
        return p;
    };
    __bf16* xb   = (__bf16*)alloc((size_t)MROWS * D_ * 2);
    __bf16* wqh  = (__bf16*)alloc((size_t)DH_ * D_ * 2);
    __bf16* wkh  = (__bf16*)alloc((size_t)DH_ * D_ * 2);
    __bf16* wv   = (__bf16*)alloc((size_t)D_ * D_ * 2);
    __bf16* wo   = (__bf16*)alloc((size_t)D_ * D_ * 2);
    __bf16* w1   = (__bf16*)alloc((size_t)F_ * D_ * 2);
    __bf16* w2   = (__bf16*)alloc((size_t)D_ * F_ * 2);
    __bf16* qh   = (__bf16*)alloc((size_t)MROWS * DH_ * 2);
    __bf16* kh   = (__bf16*)alloc((size_t)MROWS * DH_ * 2);
    __bf16* vb   = (__bf16*)alloc((size_t)MROWS * D_ * 2);
    float*  ctxf = (float*)alloc((size_t)MROWS * D_ * 4);
    __bf16* ctxb = (__bf16*)alloc((size_t)MROWS * D_ * 2);
    __bf16* hb   = (__bf16*)alloc((size_t)MROWS * F_ * 2);
    float*  yb   = (float*)alloc((size_t)MROWS * D_ * 4);

    const dim3 blk(256);

    // 1) fp32 -> bf16 conversions (Wq/Wk: only last head's 64 rows needed)
    cvt_f32_bf16<<<1024, blk, 0, stream>>>(x, xb, (size_t)MROWS * D_);
    cvt_f32_bf16<<<64,   blk, 0, stream>>>(Wq + (size_t)(H_ - 1) * DH_ * D_, wqh, (size_t)DH_ * D_);
    cvt_f32_bf16<<<64,   blk, 0, stream>>>(Wk + (size_t)(H_ - 1) * DH_ * D_, wkh, (size_t)DH_ * D_);
    cvt_f32_bf16<<<512,  blk, 0, stream>>>(Wv, wv, (size_t)D_ * D_);
    cvt_f32_bf16<<<512,  blk, 0, stream>>>(Wo, wo, (size_t)D_ * D_);
    cvt_f32_bf16<<<1024, blk, 0, stream>>>(W1, w1, (size_t)F_ * D_);
    cvt_f32_bf16<<<1024, blk, 0, stream>>>(W2, w2, (size_t)D_ * F_);

    // 2) qH = x @ WqH^T  (8192x64x1024) ; scale folded into softmax
    gemm_bf16_wmma<<<dim3((MROWS / 64) * (DH_ / 64) / WPB, 1), blk, 0, stream>>>(
        xb, wqh, nullptr, nullptr, nullptr, qh, MROWS, DH_, D_, 0, 0, 0, 0);
    //    kH = x @ WkH^T
    gemm_bf16_wmma<<<dim3((MROWS / 64) * (DH_ / 64) / WPB, 1), blk, 0, stream>>>(
        xb, wkh, nullptr, nullptr, nullptr, kh, MROWS, DH_, D_, 0, 0, 0, 0);
    // 3) v = x @ Wv^T  (8192x1024x1024)
    gemm_bf16_wmma<<<dim3((MROWS / 64) * (D_ / 64) / WPB, 1), blk, 0, stream>>>(
        xb, wv, nullptr, nullptr, nullptr, vb, MROWS, D_, D_, 0, 0, 0, 0);
    // 4) ctx = v @ Wo^T  -> fp32 (for residual/LN) + bf16 (for W1 GEMM)
    gemm_bf16_wmma<<<dim3((MROWS / 64) * (D_ / 64) / WPB, 1), blk, 0, stream>>>(
        vb, wo, nullptr, nullptr, ctxf, ctxb, MROWS, D_, D_, 0, 0, 0, 0);
    // 5) scores = qH_b @ kH_b^T per batch (4x 2048x2048x64), raw logits into d_out attn region
    gemm_bf16_wmma<<<dim3((L_ / 64) * (L_ / 64) / WPB, B_), blk, 0, stream>>>(
        qh, kh, nullptr, nullptr, attn, nullptr, L_, L_, DH_,
        (long long)L_ * DH_, (long long)L_ * DH_, (long long)L_ * L_, 0);
    // 6) h = relu(ctx @ W1^T + b1)  (8192x4096x1024)
    gemm_bf16_wmma<<<dim3((MROWS / 64) * (F_ / 64) / WPB, 1), blk, 0, stream>>>(
        ctxb, w1, b1, nullptr, nullptr, hb, MROWS, F_, D_, 0, 0, 0, 1);
    // 7) y = h @ W2^T + b2 + ctx  (8192x1024x4096), fp32
    gemm_bf16_wmma<<<dim3((MROWS / 64) * (D_ / 64) / WPB, 1), blk, 0, stream>>>(
        hb, w2, b2, ctxf, yb, nullptr, MROWS, D_, F_, 0, 0, 0, 0);

    // 8) softmax over each of the 8192 score rows (in place in d_out)
    softmax2048<<<MROWS, blk, 0, stream>>>(attn);
    // 9) LayerNorm -> out
    layernorm1024<<<MROWS, blk, 0, stream>>>(yb, lng, lnb, out);
}
